// BitLayer_9543417331797
// MI455X (gfx1250) — 8-bit matrix operand order repaired and checked by reading, compiled, not run on hardware
//
#include <hip/hip_runtime.h>
#include <stdint.h>

typedef __attribute__((ext_vector_type(8))) int v8i;

#define TF_KS0 0u
#define TF_KS1 42u
#define TF_KS2 (0x1BD11BDAu ^ 0u ^ 42u)

__device__ __forceinline__ uint32_t rotl32(uint32_t x, int d) {
  return (x << d) | (x >> (32 - d));
}
__device__ __forceinline__ void tfround(uint32_t& x0, uint32_t& x1, int r) {
  x0 += x1; x1 = rotl32(x1, r); x1 ^= x0;
}
__device__ __forceinline__ void threefry2x32(uint32_t c0, uint32_t c1,
                                             uint32_t& r0, uint32_t& r1) {
  uint32_t x0 = c0 + TF_KS0, x1 = c1 + TF_KS1;
  tfround(x0,x1,13); tfround(x0,x1,15); tfround(x0,x1,26); tfround(x0,x1, 6);
  x0 += TF_KS1; x1 += TF_KS2 + 1u;
  tfround(x0,x1,17); tfround(x0,x1,29); tfround(x0,x1,16); tfround(x0,x1,24);
  x0 += TF_KS2; x1 += TF_KS0 + 2u;
  tfround(x0,x1,13); tfround(x0,x1,15); tfround(x0,x1,26); tfround(x0,x1, 6);
  x0 += TF_KS0; x1 += TF_KS1 + 3u;
  tfround(x0,x1,17); tfround(x0,x1,29); tfround(x0,x1,16); tfround(x0,x1,24);
  x0 += TF_KS1; x1 += TF_KS2 + 4u;
  tfround(x0,x1,13); tfround(x0,x1,15); tfround(x0,x1,26); tfround(x0,x1, 6);
  x0 += TF_KS2; x1 += TF_KS0 + 5u;
  r0 = x0; r1 = x1;
}
__device__ __forceinline__ float u01(uint32_t bits) {
  return __uint_as_float((bits >> 9) | 0x3f800000u) - 1.0f;
}

__global__ __launch_bounds__(256) void pack_x_kernel(
    const int* __restrict__ x, uint8_t* __restrict__ xp) {
  uint32_t g = blockIdx.x * 256u + threadIdx.x;
  if (g >= 8388608u) return;
  uint32_t t    = g >> 13;
  uint32_t r    = g & 8191u;
  uint32_t ks   = r >> 10;
  uint32_t l2   = r & 1023u;
  uint32_t lane = l2 >> 5;
  uint32_t j    = l2 & 31u;
  uint32_t b    = lane & 15u;
  uint32_t koff = (lane >> 4) * 16u;
  uint32_t kb   = (j < 16u) ? (koff + j) : (32u + koff + (j - 16u));
  uint32_t i    = ks * 64u + kb;
  int v = x[((size_t)b * 512u + i) * 1024u + t];
  xp[g] = (uint8_t)v;
}

__global__ __launch_bounds__(32) void bitlayer_wmma_kernel(
    const uint8_t* __restrict__ xp, const float* __restrict__ kern,
    float* __restrict__ out) {
  const int m    = blockIdx.x;
  const int t    = blockIdx.y;
  const int lane = threadIdx.x;
  const int hi   = lane >> 4;
  const int mrow = lane & 15;
  const int base = hi * 8;
  const int o0   = m * 16 + mrow;
  const int o1   = o0 + 128;

  v8i C0 = {};
  v8i C1 = {};

  const uint8_t* bfrag = xp + ((size_t)t << 13) + (size_t)lane * 32;

  for (int ks = 0; ks < 8; ++ks) {
    int4 blo = *(const int4*)(bfrag + ks * 1024);
    int4 bhi = *(const int4*)(bfrag + ks * 1024 + 16);
    v8i B;
    B[0] = blo.x; B[1] = blo.y; B[2] = blo.z; B[3] = blo.w;
    B[4] = bhi.x; B[5] = bhi.y; B[6] = bhi.z; B[7] = bhi.w;

    v8i A0, A1;
#pragma unroll
    for (int run = 0; run < 4; ++run) {
      const int i0 = ks * 64 + (run >> 1) * 32 + hi * 16 + (run & 1) * 8;
      float4 p0a = *(const float4*)(kern + (size_t)o0 * 512 + i0);
      float4 p0b = *(const float4*)(kern + (size_t)o0 * 512 + i0 + 4);
      float4 p1a = *(const float4*)(kern + (size_t)o1 * 512 + i0);
      float4 p1b = *(const float4*)(kern + (size_t)o1 * 512 + i0 + 4);
      const float p0[8] = {p0a.x, p0a.y, p0a.z, p0a.w,
                           p0b.x, p0b.y, p0b.z, p0b.w};
      const float p1[8] = {p1a.x, p1a.y, p1a.z, p1a.w,
                           p1b.x, p1b.y, p1b.z, p1b.w};
      uint32_t w0lo = 0, w0hi = 0, w1lo = 0, w1hi = 0;
#pragma unroll
      for (int j = 0; j < 8; ++j) {
        uint32_t c = (((uint32_t)o0 * 512u + (uint32_t)(i0 + j)) << 10) |
                     (uint32_t)t;
        uint32_t r0, r1;
        threefry2x32(c, c + (1u << 26), r0, r1);
        uint32_t b0 = (u01(r0) < p0[j]) ? 1u : 0u;
        uint32_t b1 = (u01(r1) < p1[j]) ? 1u : 0u;
        if (j < 4) { w0lo |= b0 << (8 * j);       w1lo |= b1 << (8 * j); }
        else       { w0hi |= b0 << (8 * (j - 4)); w1hi |= b1 << (8 * (j - 4)); }
      }
      A0[2 * run] = (int)w0lo; A0[2 * run + 1] = (int)w0hi;
      A1[2 * run] = (int)w1lo; A1[2 * run + 1] = (int)w1hi;
    }

    C0 = __builtin_amdgcn_wmma_i32_16x16x64_iu8(false, A0, false, B, C0,
                                                false, false);
    C1 = __builtin_amdgcn_wmma_i32_16x16x64_iu8(false, A1, false, B, C1,
                                                false, false);
  }

#pragma unroll
  for (int r = 0; r < 8; ++r) {
    int M = r + hi * 8;
    int orow = m * 16 + M;
    size_t idx0 = (((size_t)mrow * 256 + orow) << 10) + (size_t)t;
    size_t idx1 = (((size_t)mrow * 256 + orow + 128) << 10) + (size_t)t;
    out[idx0] = (C0[r] > 0) ? 1.0f : 0.0f;
    out[idx1] = (C1[r] > 0) ? 1.0f : 0.0f;
  }
}

extern "C" void kernel_launch(void* const* d_in, const int* in_sizes, int n_in,
                              void* d_out, int out_size, void* d_ws,
                              size_t ws_size, hipStream_t stream) {
  const int*   x    = (const int*)d_in[0];
  const float* kern = (const float*)d_in[1];
  float*       out  = (float*)d_out;
  uint8_t*     xp   = (uint8_t*)d_ws;

  pack_x_kernel<<<32768, 256, 0, stream>>>(x, xp);

  dim3 grid(8, 1024);
  bitlayer_wmma_kernel<<<grid, 32, 0, stream>>>(xp, kern, out);
}
